// Qwen2MoeSparseMoeBlock_283467842487
// MI455X (gfx1250) — compile-verified, matrix operand order checked by reading, not run on hardware
//
#include <hip/hip_runtime.h>

// ---------------------------------------------------------------------------
// Qwen2 MoE sparse block for MI455X (gfx1250, wave32, WMMA).
// fp32 HBM data -> in-register cvt to bf16 -> v_wmma_f32_16x16x32_bf16.
// ---------------------------------------------------------------------------

constexpr int HID    = 2048;   // hidden size (K of first GEMMs)
constexpr int NE     = 8;      // experts
constexpr int MTOK   = 4096;   // tokens (2*2048)
constexpr int MOE_I  = 1408;   // per-expert intermediate
constexpr int SH_I   = 5632;   // shared intermediate
constexpr int NSLOT  = 2 * MTOK; // top-2 routing slots

constexpr int BM = 128, BN = 64, BK = 32, LS = 40; // LDS row stride (bf16) padded

typedef __attribute__((ext_vector_type(16))) __bf16 bf16x16;
typedef __attribute__((ext_vector_type(8)))  __bf16 bf16x8;
typedef __attribute__((ext_vector_type(8)))  float  f32x8;

__device__ __forceinline__ __bf16 f2bf(float f) {
    union { float f; unsigned u; } v; v.f = f;
    unsigned r = v.u + 0x7fffu + ((v.u >> 16) & 1u); // RNE
    union { unsigned short s; __bf16 b; } o; o.s = (unsigned short)(r >> 16);
    return o.b;
}

__device__ __forceinline__ bf16x8 pack8(float4 a, float4 b) {
    bf16x8 v;
    v[0]=f2bf(a.x); v[1]=f2bf(a.y); v[2]=f2bf(a.z); v[3]=f2bf(a.w);
    v[4]=f2bf(b.x); v[5]=f2bf(b.y); v[6]=f2bf(b.z); v[7]=f2bf(b.w);
    return v;
}

// A/B fragment for 16x16x32 bf16 WMMA. Lane L (L&15 = row/col index) holds
// 8 contiguous bf16 at k = (L>>4)*8 and 8 more at k+16  (ISA 16-bit A layout).
__device__ __forceinline__ bf16x16 load_frag(const __bf16* p) {
    union { bf16x16 v; bf16x8 h[2]; } u;
    u.h[0] = *(const bf16x8*)(p);
    u.h[1] = *(const bf16x8*)(p + 16);
    return u.v;
}

__device__ __forceinline__ f32x8 wmma_bf16(bf16x16 a, bf16x16 b, f32x8 c) {
    return __builtin_amdgcn_wmma_f32_16x16x32_bf16(false, a, false, b,
                                                   (short)0, c, false, false);
}

// ---------------------------------------------------------------------------
// Kernel 1: router. One wave32 per token: 9 dot products (8 experts + shared
// scalar gate), softmax fp32, top-2 (NO renorm), sigmoid gate, expert counts.
// ---------------------------------------------------------------------------
__global__ __launch_bounds__(256)
void router_kernel(const float* __restrict__ X, const float* __restrict__ Gw,
                   const float* __restrict__ SGw,
                   int* __restrict__ tk_idx, float* __restrict__ tk_w,
                   float* __restrict__ sig, int* __restrict__ counts)
{
    const int wv   = threadIdx.x >> 5;
    const int lane = threadIdx.x & 31;
    const int tok  = blockIdx.x * 8 + wv;
    const float* x = X + (size_t)tok * HID;

    float acc[9];
    #pragma unroll
    for (int e = 0; e < 9; ++e) acc[e] = 0.0f;

    for (int d = lane * 4; d < HID; d += 128) {
        float4 xv = *(const float4*)(x + d);
        #pragma unroll
        for (int e = 0; e < 8; ++e) {
            float4 w4 = *(const float4*)(Gw + e * HID + d);
            acc[e] += xv.x*w4.x + xv.y*w4.y + xv.z*w4.z + xv.w*w4.w;
        }
        float4 s4 = *(const float4*)(SGw + d);
        acc[8] += xv.x*s4.x + xv.y*s4.y + xv.z*s4.z + xv.w*s4.w;
    }
    #pragma unroll
    for (int off = 16; off; off >>= 1)
        #pragma unroll
        for (int e = 0; e < 9; ++e)
            acc[e] += __shfl_xor(acc[e], off, 32);

    if (lane == 0) {
        float mx = acc[0];
        #pragma unroll
        for (int e = 1; e < 8; ++e) mx = fmaxf(mx, acc[e]);
        float p[8], s = 0.0f;
        #pragma unroll
        for (int e = 0; e < 8; ++e) { p[e] = __expf(acc[e] - mx); s += p[e]; }
        float inv = 1.0f / s;
        #pragma unroll
        for (int e = 0; e < 8; ++e) p[e] *= inv;
        int i1 = 0;
        #pragma unroll
        for (int e = 1; e < 8; ++e) if (p[e] > p[i1]) i1 = e;
        int i2 = (i1 == 0) ? 1 : 0;
        #pragma unroll
        for (int e = 0; e < 8; ++e) if (e != i1 && p[e] > p[i2]) i2 = e;
        tk_idx[tok*2+0] = i1;  tk_w[tok*2+0] = p[i1];
        tk_idx[tok*2+1] = i2;  tk_w[tok*2+1] = p[i2];
        sig[tok] = 1.0f / (1.0f + __expf(-acc[8]));
        atomicAdd(&counts[i1], 1);
        atomicAdd(&counts[i2], 1);
    }
}

// Kernel 2: 8-entry exclusive scan.
__global__ void scan_kernel(const int* __restrict__ counts, int* __restrict__ offsets) {
    if (threadIdx.x == 0) {
        int s = 0;
        for (int e = 0; e < NE; ++e) { offsets[e] = s; s += counts[e]; }
    }
}

// Kernel 3: compact per-expert token lists.
__global__ __launch_bounds__(256)
void assign_kernel(const int* __restrict__ tk_idx, const float* __restrict__ tk_w,
                   const int* __restrict__ offsets, int* __restrict__ cursors,
                   int* __restrict__ slot_tok, float* __restrict__ slot_w)
{
    int t = blockIdx.x * blockDim.x + threadIdx.x;
    if (t >= MTOK) return;
    #pragma unroll
    for (int k = 0; k < 2; ++k) {
        int e = tk_idx[t*2+k];
        int pos = offsets[e] + atomicAdd(&cursors[e], 1);
        slot_tok[pos] = t;
        slot_w[pos]   = tk_w[t*2+k];
    }
}

// ---------------------------------------------------------------------------
// Kernel 4: fused gate_up GEMM + silu*mul -> bf16 intermediate.
// C_gate = A @ Wg^T, C_up = A @ Wu^T share the A tile. A is fp32 (optionally
// gathered by slot list), converted to bf16 while staging into LDS.
// Grid: (I/BN, ceil(M/BM), experts).  K = HID.
// ---------------------------------------------------------------------------
template<bool GATHER>
__global__ __launch_bounds__(256)
void gateup_kernel(const float* __restrict__ X, const float* __restrict__ Wbase,
                   int I, __bf16* __restrict__ Hout,
                   const int* __restrict__ slot_tok,
                   const int* __restrict__ ex_count, const int* __restrict__ ex_offset)
{
    __shared__ __bf16 Al[BM * LS];
    __shared__ __bf16 Bg[BN * LS];
    __shared__ __bf16 Bu[BN * LS];

    const int e = blockIdx.z;
    int cnt = MTOK, out_base = 0;
    const int* toks = nullptr;
    if constexpr (GATHER) {
        cnt      = ex_count[e];
        out_base = ex_offset[e];
        toks     = slot_tok + out_base;
    }
    const int m0 = blockIdx.y * BM;
    if (m0 >= cnt) return;
    const int n0 = blockIdx.x * BN;

    const float* Wg = Wbase + (size_t)e * (size_t)(2 * I) * HID;
    const float* Wu = Wg + (size_t)I * HID;

    const int tid = threadIdx.x;
    // A staging: 2 threads per row, 16 floats each.
    const int ar = tid >> 1;
    const int ac = (tid & 1) * 16;
    int arow = m0 + ar;
    bool avalid = (arow < cnt);
    const float* Arow;
    if constexpr (GATHER) Arow = X + (size_t)(avalid ? toks[arow] : toks[0]) * HID;
    else                  Arow = X + (size_t)(avalid ? arow : m0) * HID;
    // B staging: 4 threads per row, 8 floats each.
    const int br = tid >> 2;
    const int bc = (tid & 3) * 8;
    const float* Bgrow = Wg + (size_t)(n0 + br) * HID;
    const float* Burow = Wu + (size_t)(n0 + br) * HID;

    __bf16* sA  = &Al[ar * LS + ac];
    __bf16* sBg = &Bg[br * LS + bc];
    __bf16* sBu = &Bu[br * LS + bc];

    const int lane = tid & 31;
    const int wv   = tid >> 5;
    const int wm   = wv & 3;     // 4 row-slices of 32
    const int wn   = wv >> 2;    // 2 col-slices of 32
    const int fr   = lane & 15;
    const int fk   = (lane >> 4) * 8;
    const int mh   = (lane >> 4) * 8;

    const __bf16* pA[2]; const __bf16* pG[2]; const __bf16* pU[2];
    #pragma unroll
    for (int i = 0; i < 2; ++i) pA[i] = &Al[(wm*32 + i*16 + fr) * LS + fk];
    #pragma unroll
    for (int j = 0; j < 2; ++j) {
        pG[j] = &Bg[(wn*32 + j*16 + fr) * LS + fk];
        pU[j] = &Bu[(wn*32 + j*16 + fr) * LS + fk];
    }

    f32x8 accg[2][2], accu[2][2];
    const f32x8 zero = {0.f,0.f,0.f,0.f,0.f,0.f,0.f,0.f};
    #pragma unroll
    for (int i = 0; i < 2; ++i)
        #pragma unroll
        for (int j = 0; j < 2; ++j) { accg[i][j] = zero; accu[i][j] = zero; }

    for (int k0 = 0; k0 < HID; k0 += BK) {
        const float4* ga = (const float4*)(Arow + k0 + ac);
        float4 a0 = ga[0], a1 = ga[1], a2 = ga[2], a3 = ga[3];
        const float4* gg = (const float4*)(Bgrow + k0 + bc);
        float4 g0 = gg[0], g1 = gg[1];
        const float4* gu = (const float4*)(Burow + k0 + bc);
        float4 u0 = gu[0], u1 = gu[1];
        if (k0 + BK < HID) {  // stream-in hints for next weight tile
            __builtin_prefetch(Bgrow + k0 + BK + bc, 0, 0);
            __builtin_prefetch(Burow + k0 + BK + bc, 0, 0);
        }
        __syncthreads();
        *(bf16x8*)(sA)     = pack8(a0, a1);
        *(bf16x8*)(sA + 8) = pack8(a2, a3);
        *(bf16x8*)(sBg)    = pack8(g0, g1);
        *(bf16x8*)(sBu)    = pack8(u0, u1);
        __syncthreads();

        bf16x16 af0 = load_frag(pA[0]);
        bf16x16 af1 = load_frag(pA[1]);
        bf16x16 bg0 = load_frag(pG[0]);
        bf16x16 bg1 = load_frag(pG[1]);
        bf16x16 bu0 = load_frag(pU[0]);
        bf16x16 bu1 = load_frag(pU[1]);
        accg[0][0] = wmma_bf16(af0, bg0, accg[0][0]);
        accg[0][1] = wmma_bf16(af0, bg1, accg[0][1]);
        accg[1][0] = wmma_bf16(af1, bg0, accg[1][0]);
        accg[1][1] = wmma_bf16(af1, bg1, accg[1][1]);
        accu[0][0] = wmma_bf16(af0, bu0, accu[0][0]);
        accu[0][1] = wmma_bf16(af0, bu1, accu[0][1]);
        accu[1][0] = wmma_bf16(af1, bu0, accu[1][0]);
        accu[1][1] = wmma_bf16(af1, bu1, accu[1][1]);
    }

    // Epilogue: h = silu(gate) * up, store bf16. D layout: lane&15 = column,
    // VGPR v = row ((lane>>4)*8 + v) within the 16x16 tile.
    #pragma unroll
    for (int i = 0; i < 2; ++i) {
        #pragma unroll
        for (int v = 0; v < 8; ++v) {
            int m = m0 + wm*32 + i*16 + mh + v;
            if (m < cnt) {
                size_t rowoff = (size_t)(out_base + m) * I;
                #pragma unroll
                for (int j = 0; j < 2; ++j) {
                    int n = n0 + wn*32 + j*16 + fr;
                    float g  = accg[i][j][v];
                    float uu = accu[i][j][v];
                    float hv = (g / (1.0f + __expf(-g))) * uu;
                    Hout[rowoff + n] = f2bf(hv);
                }
            }
        }
    }
}

// ---------------------------------------------------------------------------
// Kernel 5: down-proj GEMM. A is bf16 intermediate [rows, I], W fp32 [HID, I].
// EXPERT: atomically accumulates routing_w * C into d_out[token].
// !EXPERT (shared): writes sigmoid_gate[m] * C into d_out (initializes it).
// Grid: (HID/BN, ceil(M/BM), experts).
// ---------------------------------------------------------------------------
template<bool EXPERT>
__global__ __launch_bounds__(256)
void down_kernel(const __bf16* __restrict__ Hin, const float* __restrict__ Wbase,
                 int I, float* __restrict__ Out,
                 const float* __restrict__ sig,
                 const int* __restrict__ slot_tok, const float* __restrict__ slot_w,
                 const int* __restrict__ ex_count, const int* __restrict__ ex_offset)
{
    __shared__ __bf16 Al[BM * LS];
    __shared__ __bf16 Bl[BN * LS];

    const int e = blockIdx.z;
    int cnt = MTOK, base = 0;
    if constexpr (EXPERT) { cnt = ex_count[e]; base = ex_offset[e]; }
    const int m0 = blockIdx.y * BM;
    if (m0 >= cnt) return;
    const int n0 = blockIdx.x * BN;

    const float* W = Wbase + (size_t)e * (size_t)HID * I;

    const int tid = threadIdx.x;
    const int ar = tid >> 1;
    const int ac = (tid & 1) * 16;
    int arow = m0 + ar;
    if (arow >= cnt) arow = m0;                 // clamp inside segment
    const __bf16* Arow = Hin + (size_t)(base + arow) * I;
    const int br = tid >> 2;
    const int bc = (tid & 3) * 8;
    const float* Brow = W + (size_t)(n0 + br) * I;

    __bf16* sA = &Al[ar * LS + ac];
    __bf16* sB = &Bl[br * LS + bc];

    const int lane = tid & 31;
    const int wv   = tid >> 5;
    const int wm   = wv & 3;
    const int wn   = wv >> 2;
    const int fr   = lane & 15;
    const int fk   = (lane >> 4) * 8;
    const int mh   = (lane >> 4) * 8;

    const __bf16* pA[2]; const __bf16* pB[2];
    #pragma unroll
    for (int i = 0; i < 2; ++i) pA[i] = &Al[(wm*32 + i*16 + fr) * LS + fk];
    #pragma unroll
    for (int j = 0; j < 2; ++j) pB[j] = &Bl[(wn*32 + j*16 + fr) * LS + fk];

    f32x8 acc[2][2];
    const f32x8 zero = {0.f,0.f,0.f,0.f,0.f,0.f,0.f,0.f};
    #pragma unroll
    for (int i = 0; i < 2; ++i)
        #pragma unroll
        for (int j = 0; j < 2; ++j) acc[i][j] = zero;

    for (int k0 = 0; k0 < I; k0 += BK) {
        const __bf16* gsrc = Arow + k0 + ac;
        bf16x8 alo = *(const bf16x8*)(gsrc);
        bf16x8 ahi = *(const bf16x8*)(gsrc + 8);
        const float4* gb = (const float4*)(Brow + k0 + bc);
        float4 b0 = gb[0], b1 = gb[1];
        if (k0 + BK < I) __builtin_prefetch(Brow + k0 + BK + bc, 0, 0);
        __syncthreads();
        *(bf16x8*)(sA)     = alo;
        *(bf16x8*)(sA + 8) = ahi;
        *(bf16x8*)(sB)     = pack8(b0, b1);
        __syncthreads();

        bf16x16 af0 = load_frag(pA[0]);
        bf16x16 af1 = load_frag(pA[1]);
        bf16x16 bf0 = load_frag(pB[0]);
        bf16x16 bf1 = load_frag(pB[1]);
        acc[0][0] = wmma_bf16(af0, bf0, acc[0][0]);
        acc[0][1] = wmma_bf16(af0, bf1, acc[0][1]);
        acc[1][0] = wmma_bf16(af1, bf0, acc[1][0]);
        acc[1][1] = wmma_bf16(af1, bf1, acc[1][1]);
    }

    #pragma unroll
    for (int i = 0; i < 2; ++i) {
        #pragma unroll
        for (int v = 0; v < 8; ++v) {
            int m = m0 + wm*32 + i*16 + mh + v;
            if (m < cnt) {
                if constexpr (EXPERT) {
                    int slot   = base + m;
                    int tokn   = slot_tok[slot];
                    float w    = slot_w[slot];
                    float* row = Out + (size_t)tokn * HID;
                    #pragma unroll
                    for (int j = 0; j < 2; ++j) {
                        int n = n0 + wn*32 + j*16 + fr;
                        atomicAdd(row + n, w * acc[i][j][v]);
                    }
                } else {
                    float sg   = sig[m];
                    float* row = Out + (size_t)m * HID;
                    #pragma unroll
                    for (int j = 0; j < 2; ++j) {
                        int n = n0 + wn*32 + j*16 + fr;
                        row[n] = sg * acc[i][j][v];
                    }
                }
            }
        }
    }
}

// ---------------------------------------------------------------------------
// Host launcher. Workspace layout (~69.4 MB):
//   h_shared bf16[4096*5632] | h_exp bf16[8192*1408] | slot_tok i32[8192] |
//   slot_w f32[8192] | sig f32[4096] | tk_idx i32[8192] | tk_w f32[8192] |
//   counts/offsets/cursors i32[8] each.
// ---------------------------------------------------------------------------
extern "C" void kernel_launch(void* const* d_in, const int* in_sizes, int n_in,
                              void* d_out, int out_size, void* d_ws, size_t ws_size,
                              hipStream_t stream) {
    const float* x    = (const float*)d_in[0];
    const float* gw   = (const float*)d_in[1];
    const float* sgw  = (const float*)d_in[2];
    const float* sguw = (const float*)d_in[3];
    const float* sdw  = (const float*)d_in[4];
    const float* w13  = (const float*)d_in[5];
    const float* w2   = (const float*)d_in[6];
    float* out = (float*)d_out;

    __bf16* h_sh     = (__bf16*)d_ws;
    __bf16* h_ex     = h_sh + (size_t)MTOK * SH_I;
    int*    slot_tok = (int*)(h_ex + (size_t)NSLOT * MOE_I);
    float*  slot_w   = (float*)(slot_tok + NSLOT);
    float*  sig      = slot_w + NSLOT;
    int*    tk_idx   = (int*)(sig + MTOK);
    float*  tk_w     = (float*)(tk_idx + 2 * MTOK);
    int*    counts   = (int*)(tk_w + 2 * MTOK);
    int*    offsets  = counts + NE;
    int*    cursors  = offsets + NE;

    hipMemsetAsync(counts, 0, 3 * NE * sizeof(int), stream);

    router_kernel<<<MTOK / 8, 256, 0, stream>>>(x, gw, sgw, tk_idx, tk_w, sig, counts);
    scan_kernel<<<1, 32, 0, stream>>>(counts, offsets);
    assign_kernel<<<MTOK / 256, 256, 0, stream>>>(tk_idx, tk_w, offsets, cursors,
                                                  slot_tok, slot_w);

    // Shared expert: gate_up (+silu*mul) then down (writes d_out * sigmoid gate).
    gateup_kernel<false><<<dim3(SH_I / BN, MTOK / BM, 1), 256, 0, stream>>>(
        x, sguw, SH_I, h_sh, nullptr, nullptr, nullptr);
    down_kernel<false><<<dim3(HID / BN, MTOK / BM, 1), 256, 0, stream>>>(
        h_sh, sdw, SH_I, out, sig, nullptr, nullptr, nullptr, nullptr);

    // Routed experts (grouped GEMM over compact slot lists; blocks early-exit).
    gateup_kernel<true><<<dim3(MOE_I / BN, MTOK / BM, NE), 256, 0, stream>>>(
        x, w13, MOE_I, h_ex, slot_tok, counts, offsets);
    down_kernel<true><<<dim3(HID / BN, MTOK / BM, NE), 256, 0, stream>>>(
        h_ex, w2, MOE_I, out, nullptr, slot_tok, slot_w, counts, offsets);
}
